// CFConv_87677462380692
// MI455X (gfx1250) — matrix operand order repaired and checked by reading, not compiled, not run on hardware
//
#include <hip/hip_runtime.h>
#include <hip/hip_bf16.h>

typedef __attribute__((ext_vector_type(16))) __bf16 v16bf;
typedef __attribute__((ext_vector_type(8)))  float  v8f;

#define F_DIM   128
#define RBF_DIM 64
#define NBR     32

// f32 -> bf16 round-to-nearest-even, pure bit manipulation (no __bf16 math)
static __device__ __forceinline__ unsigned short f2bf(float f) {
    unsigned u = __float_as_uint(f);
    u += 0x7FFFu + ((u >> 16) & 1u);
    return (unsigned short)(u >> 16);
}
static __device__ __forceinline__ unsigned int pack_bf2(float lo, float hi) {
    return ((unsigned int)f2bf(hi) << 16) | (unsigned int)f2bf(lo);
}

union AFrag {
    v16bf          v;
    unsigned short h[16];
    unsigned int   w[8];
    __device__ AFrag() {}
};

static __device__ __forceinline__ float softplus_f(float t) {
    return fmaxf(t, 0.0f) + __logf(1.0f + __expf(-fabsf(t)));
}

// ---------------------------------------------------------------------------
// Prep: pre-swizzle w1 [64x128] and w2 [128x128] (f32, row-major [k][n]) into
// bf16 WMMA B-fragment order in d_ws.
//   B-frag (16x16x32 bf16): lane = {hi=lane>>4, n=lane&15},
//   element e (0..15): K = 32*s + (e&7) + 8*hi + 16*(e>>3),  N = c*16 + n
//   storage index: (((s*8 + c)*32 + lane)*16 + e)
// ---------------------------------------------------------------------------
__global__ void cfconv_prep(const float* __restrict__ w1,
                            const float* __restrict__ w2,
                            unsigned short* __restrict__ w1f,
                            unsigned short* __restrict__ w2f) {
    const int TOT1 = 2 * 8 * 32 * 16;   // 8192
    const int TOT2 = 4 * 8 * 32 * 16;   // 16384
    for (int i = threadIdx.x + blockIdx.x * blockDim.x; i < TOT1 + TOT2;
         i += blockDim.x * gridDim.x) {
        const float* w;
        unsigned short* dst;
        int idx;
        if (i < TOT1) { w = w1; dst = w1f; idx = i; }
        else          { w = w2; dst = w2f; idx = i - TOT1; }
        int e    = idx & 15;
        int lane = (idx >> 4) & 31;
        int c    = (idx >> 9) & 7;
        int s    = idx >> 12;
        int k = 32 * s + (e & 7) + 8 * (lane >> 4) + 16 * (e >> 3);
        int n = (c << 4) + (lane & 15);
        dst[idx] = f2bf(w[k * F_DIM + n]);
    }
}

// ---------------------------------------------------------------------------
// Fused CFConv: one workgroup (256 thr = 8 waves) per atom.
//   Stage rbf tile (32x64 f32 -> bf16 LDS) coalesced once per block.
//   wave w owns column tile c=w; both row tiles r=0,1 via dual accumulators.
//   A-fragments = two ds b128 loads each (16-bit A layout is two contiguous
//   8-element K-runs per lane). B-fragments preswizzled in d_ws.
// ---------------------------------------------------------------------------
__global__ __launch_bounds__(256) void cfconv_main(
    const float* __restrict__ x, const float* __restrict__ rbf,
    const long long* __restrict__ nbrs,
    const float* __restrict__ b1, const float* __restrict__ b2,
    const unsigned short* __restrict__ w1f_u16,
    const unsigned short* __restrict__ w2f_u16,
    float* __restrict__ out) {

    const int a    = blockIdx.x;
    const int tid  = threadIdx.x;
    const int wave = tid >> 5;      // 0..7 == column tile
    const int lane = tid & 31;
    const int hi   = lane >> 4;     // 0/1 half-wave
    const int lm   = lane & 15;

    const v16bf* __restrict__ w1f = (const v16bf*)w1f_u16;
    const v16bf* __restrict__ w2f = (const v16bf*)w2f_u16;

    __shared__ int            s_nbr[NBR];
    __shared__ unsigned short s_rbf[NBR][RBF_DIM + 8]; // bf16, 144B rows (16B mult)
    __shared__ unsigned short s_h[NBR][F_DIM + 8];     // bf16, 272B rows (16B mult)

    if (tid < NBR) s_nbr[tid] = (int)nbrs[(long long)a * NBR + tid];

    // ---- stage rbf tile -> bf16 LDS: 2048 floats, 8 per thread, coalesced ----
    {
        int row = tid >> 3;
        int col = (tid & 7) << 3;
        const float* src = rbf + ((size_t)a * NBR + row) * RBF_DIM + col;
        float4 f0 = *(const float4*)src;
        float4 f1 = *(const float4*)(src + 4);
        uint4 p;
        p.x = pack_bf2(f0.x, f0.y);
        p.y = pack_bf2(f0.z, f0.w);
        p.z = pack_bf2(f1.x, f1.y);
        p.w = pack_bf2(f1.z, f1.w);
        *(uint4*)&s_rbf[row][col] = p;
    }
    __syncthreads();

    const int   ncol  = wave * 16 + lm;   // this lane's output column
    const float bias1 = b1[ncol];
    const float bias2 = b2[ncol];

    // ---------------- GEMM1: h = softplus(rbf @ w1 + b1) ----------------
    {
        v8f acc0 = {}, acc1 = {};
#pragma unroll
        for (int s = 0; s < 2; ++s) {
            v16bf bf = w1f[(s * 8 + wave) * 32 + lane];
            AFrag a0, a1;
            const unsigned short* r0 = &s_rbf[lm][32 * s + 8 * hi];
            const unsigned short* r1 = &s_rbf[16 + lm][32 * s + 8 * hi];
            ((uint4*)a0.w)[0] = *(const uint4*)r0;        // K = 0..7   (+8*hi)
            ((uint4*)a0.w)[1] = *(const uint4*)(r0 + 16); // K = 16..23 (+8*hi)
            ((uint4*)a1.w)[0] = *(const uint4*)r1;
            ((uint4*)a1.w)[1] = *(const uint4*)(r1 + 16);
            acc0 = __builtin_amdgcn_wmma_f32_16x16x32_bf16(
                false, a0.v, false, bf, (short)0, acc0, false, false);
            acc1 = __builtin_amdgcn_wmma_f32_16x16x32_bf16(
                false, a1.v, false, bf, (short)0, acc1, false, false);
        }
        // epilogue: +b1, softplus, bf16 -> LDS
        // C layout: VGPR v -> row (v + 8*hi), col lm (within tile)
#pragma unroll
        for (int v = 0; v < 8; ++v) {
            s_h[v + 8 * hi][ncol]      = f2bf(softplus_f(acc0[v] + bias1));
            s_h[16 + v + 8 * hi][ncol] = f2bf(softplus_f(acc1[v] + bias1));
        }
    }
    __syncthreads();

    // ------- GEMM2 + gather-dot: out = sum_m x[nbr[m]] * (h@w2 + b2) -------
    {
        v8f acc0 = {}, acc1 = {};
#pragma unroll
        for (int s = 0; s < 4; ++s) {
            v16bf bf = w2f[(s * 8 + wave) * 32 + lane];
            AFrag a0, a1;
            const unsigned short* r0 = &s_h[lm][32 * s + 8 * hi];
            const unsigned short* r1 = &s_h[16 + lm][32 * s + 8 * hi];
            ((uint4*)a0.w)[0] = *(const uint4*)r0;
            ((uint4*)a0.w)[1] = *(const uint4*)(r0 + 16);
            ((uint4*)a1.w)[0] = *(const uint4*)r1;
            ((uint4*)a1.w)[1] = *(const uint4*)(r1 + 16);
            acc0 = __builtin_amdgcn_wmma_f32_16x16x32_bf16(
                false, a0.v, false, bf, (short)0, acc0, false, false);
            acc1 = __builtin_amdgcn_wmma_f32_16x16x32_bf16(
                false, a1.v, false, bf, (short)0, acc1, false, false);
        }
        // epilogue: +b2, multiply gathered x, reduce over this lane's 16 rows
        float partial = 0.0f;
#pragma unroll
        for (int v = 0; v < 8; ++v) {
            int m0 = v + 8 * hi;           // neighbor slot, row tile 0
            int m1 = 16 + v + 8 * hi;      // neighbor slot, row tile 1
            float x0 = x[(size_t)s_nbr[m0] * F_DIM + ncol];
            float x1 = x[(size_t)s_nbr[m1] * F_DIM + ncol];
            partial = fmaf(acc0[v] + bias2, x0, partial);
            partial = fmaf(acc1[v] + bias2, x1, partial);
        }
        // lanes l and l+16 hold the same column; combine halves (wave32)
        partial += __shfl_xor(partial, 16, 32);
        if (hi == 0) out[(size_t)a * F_DIM + ncol] = partial;
    }
}

extern "C" void kernel_launch(void* const* d_in, const int* in_sizes, int n_in,
                              void* d_out, int out_size, void* d_ws, size_t ws_size,
                              hipStream_t stream) {
    const float*     x    = (const float*)d_in[0];
    const float*     rbf  = (const float*)d_in[1];
    const long long* nbrs = (const long long*)d_in[2];   // int64 neighbors
    const float*     w1   = (const float*)d_in[3];
    const float*     b1   = (const float*)d_in[4];
    const float*     w2   = (const float*)d_in[5];
    const float*     b2   = (const float*)d_in[6];
    float*           out  = (float*)d_out;

    const int N = in_sizes[0] / F_DIM;   // 20000

    unsigned short* w1f = (unsigned short*)d_ws;         // 8192 u16
    unsigned short* w2f = w1f + 2 * 8 * 32 * 16;         // 16384 u16 (48 KB total)

    cfconv_prep<<<96, 256, 0, stream>>>(w1, w2, w1f, w2f);
    cfconv_main<<<N, 256, 0, stream>>>(x, rbf, nbrs, b1, b2, w1f, w2f, out);
}
